// TransformerEncoderLayer_20839181320780
// MI455X (gfx1250) — compile-verified, matrix operand order checked by reading, not run on hardware
//
#include <hip/hip_runtime.h>
#include <hip/hip_bf16.h>
#include <math.h>

// ---------- types ----------
typedef __bf16 bf16_t;
typedef __attribute__((ext_vector_type(16))) __bf16 v16bf;
typedef __attribute__((ext_vector_type(8)))  __bf16 v8bf;
typedef __attribute__((ext_vector_type(8)))  float  v8f;
typedef __attribute__((ext_vector_type(4)))  float  v4f;

#define B_  8
#define N_  1024
#define D_  512
#define H_  8
#define DH_ 64
#define DFF_ 2048
#define MTOT (B_*N_)   // 8192 rows

// ---------- WMMA helpers ----------
// A-fragment (16x32 bf16, row-major source):
// lane L: row r = L&15, half = L>>4.  elements 0..7  <- src[half*8 + j]
//                                     elements 8..15 <- src[16 + half*8 + j]
// B-fragment for D = A x B is the identical pattern applied to B^T row-major.
__device__ inline v16bf load_frag(const bf16_t* __restrict__ p, int half) {
  union { v16bf v; struct { v8bf lo; v8bf hi; } s; } u;
  u.s.lo = *reinterpret_cast<const v8bf*>(p + half * 8);
  u.s.hi = *reinterpret_cast<const v8bf*>(p + 16 + half * 8);
  return u.v;
}

__device__ inline v8f wmma_bf16(v16bf a, v16bf b, v8f c) {
  return __builtin_amdgcn_wmma_f32_16x16x32_bf16(
      /*neg_a=*/false, a, /*neg_b=*/false, b,
      /*c_mod=*/(short)0, c, /*reuse_a=*/false, /*reuse_b=*/false);
}

// ---------- weight transpose + bf16 convert: Wt[n*K + k] = W[k*N + n] ----------
__global__ void wtrans_kernel(const float* __restrict__ W, bf16_t* __restrict__ Wt,
                              int K, int N) {
  int idx = blockIdx.x * blockDim.x + threadIdx.x;
  if (idx >= K * N) return;
  int n = idx / K;
  int k = idx - n * K;
  Wt[(size_t)n * K + k] = (__bf16)W[(size_t)k * N + n];
}

// ---------- LayerNorm (one wave per row of 512), optional +PE stream ----------
template <bool HAS_PE>
__global__ void ln_kernel(const float* __restrict__ x, const float* __restrict__ pe,
                          const float* __restrict__ g, const float* __restrict__ bta,
                          bf16_t* __restrict__ x2, bf16_t* __restrict__ qkin) {
  const int wave = threadIdx.x >> 5;
  const int lane = threadIdx.x & 31;
  const int row  = blockIdx.x * (blockDim.x >> 5) + wave;
  const float* xr = x + (size_t)row * D_ + lane * 16;
  float v[16];
#pragma unroll
  for (int i = 0; i < 4; ++i) {
    v4f t = *reinterpret_cast<const v4f*>(xr + i * 4);
    v[4*i+0] = t[0]; v[4*i+1] = t[1]; v[4*i+2] = t[2]; v[4*i+3] = t[3];
  }
  float s = 0.f;
#pragma unroll
  for (int i = 0; i < 16; ++i) s += v[i];
#pragma unroll
  for (int off = 16; off >= 1; off >>= 1) s += __shfl_xor(s, off, 32);
  const float mu = s * (1.0f / (float)D_);
  float vs = 0.f;
#pragma unroll
  for (int i = 0; i < 16; ++i) { float d = v[i] - mu; vs += d * d; }
#pragma unroll
  for (int off = 16; off >= 1; off >>= 1) vs += __shfl_xor(vs, off, 32);
  const float rstd = rsqrtf(vs * (1.0f / (float)D_) + 1e-5f);

  const int colb = lane * 16;
  float y[16];
#pragma unroll
  for (int i = 0; i < 16; ++i)
    y[i] = (v[i] - mu) * rstd * g[colb + i] + bta[colb + i];

  union { v8bf v; __bf16 e[8]; } p0, p1;
#pragma unroll
  for (int i = 0; i < 8; ++i) { p0.e[i] = (__bf16)y[i]; p1.e[i] = (__bf16)y[i + 8]; }
  bf16_t* xo = x2 + (size_t)row * D_ + colb;
  *reinterpret_cast<v8bf*>(xo)     = p0.v;
  *reinterpret_cast<v8bf*>(xo + 8) = p1.v;

  if (HAS_PE) {
    const float* pr = pe + (size_t)row * D_ + colb;
    float z[16];
#pragma unroll
    for (int i = 0; i < 4; ++i) {
      v4f t = *reinterpret_cast<const v4f*>(pr + i * 4);
      z[4*i+0] = y[4*i+0] + t[0]; z[4*i+1] = y[4*i+1] + t[1];
      z[4*i+2] = y[4*i+2] + t[2]; z[4*i+3] = y[4*i+3] + t[3];
    }
#pragma unroll
    for (int i = 0; i < 8; ++i) { p0.e[i] = (__bf16)z[i]; p1.e[i] = (__bf16)z[i + 8]; }
    bf16_t* qo = qkin + (size_t)row * D_ + colb;
    *reinterpret_cast<v8bf*>(qo)     = p0.v;
    *reinterpret_cast<v8bf*>(qo + 8) = p1.v;
  }
}

// ---------- 64x64-tile GEMM: out = A[MxKK] @ W[KKxNN] + bias, A/W bf16, acc f32 ----------
// Wt is W transposed: Wt[n*KK + k].  One wave per 64x64 tile.
// MODE 0: bf16 store           (Q/K projections)
// MODE 1: bf16 store, scattered as Vt[b][h][d][n]  (V projection)
// MODE 2: f32 store + residual (attn proj, MLP2)
// MODE 3: bf16 store + exact GELU (MLP1)
template <int NN, int KK, int MODE>
__global__ void __launch_bounds__(32)
gemm_wmma(const bf16_t* __restrict__ A, const bf16_t* __restrict__ Wt,
          const float* __restrict__ bias, const float* __restrict__ resid,
          bf16_t* __restrict__ outB, float* __restrict__ outF) {
  const int lane = threadIdx.x;
  const int r = lane & 15, half = lane >> 4;
  const int mb = blockIdx.y * 64, nb = blockIdx.x * 64;

  // per-lane row base pointers, hoisted out of the k-loop
  const bf16_t* ap[4];
  const bf16_t* bp[4];
#pragma unroll
  for (int i = 0; i < 4; ++i) {
    ap[i] = A  + (size_t)(mb + i * 16 + r) * KK;
    bp[i] = Wt + (size_t)(nb + i * 16 + r) * KK;
  }

  v8f acc[4][4];
#pragma unroll
  for (int i = 0; i < 4; ++i)
#pragma unroll
    for (int j = 0; j < 4; ++j)
#pragma unroll
      for (int e = 0; e < 8; ++e) acc[i][j][e] = 0.f;

  for (int k0 = 0; k0 < KK; k0 += 32) {
    // issue all 16 loads first -> one clause, one wait, then 16 back-to-back WMMAs
    v16bf a[4], bf[4];
#pragma unroll
    for (int i = 0; i < 4; ++i) a[i]  = load_frag(ap[i] + k0, half);
#pragma unroll
    for (int j = 0; j < 4; ++j) bf[j] = load_frag(bp[j] + k0, half);
#pragma unroll
    for (int j = 0; j < 4; ++j)
#pragma unroll
      for (int i = 0; i < 4; ++i)
        acc[i][j] = wmma_bf16(a[i], bf[j], acc[i][j]);
  }

  // C layout: lane holds column n = nb + j*16 + r; VGPR e holds row m = mb + i*16 + 8*half + e
#pragma unroll
  for (int j = 0; j < 4; ++j) {
    const int n  = nb + j * 16 + r;
    const float bj = bias[n];
#pragma unroll
    for (int i = 0; i < 4; ++i) {
      const int m0 = mb + i * 16 + 8 * half;
      if (MODE == 0) {
#pragma unroll
        for (int e = 0; e < 8; ++e)
          outB[(size_t)(m0 + e) * NN + n] = (__bf16)(acc[i][j][e] + bj);
      } else if (MODE == 1) {
        // V-transposed store: Vt[((b*H + h)*64 + d)*1024 + ntok], 8 consecutive ntok
        const int bidx = (mb + i * 16) >> 10;
        const int nt0  = ((mb + i * 16) & 1023) + 8 * half;
        const int hh = n >> 6, d = n & 63;
        union { v8bf v; __bf16 e[8]; } pk;
#pragma unroll
        for (int e = 0; e < 8; ++e) pk.e[e] = (__bf16)(acc[i][j][e] + bj);
        *reinterpret_cast<v8bf*>(outB + ((size_t)((bidx * H_ + hh) * DH_ + d)) * N_ + nt0) = pk.v;
      } else if (MODE == 2) {
#pragma unroll
        for (int e = 0; e < 8; ++e) {
          const size_t idx = (size_t)(m0 + e) * NN + n;
          outF[idx] = acc[i][j][e] + bj + resid[idx];
        }
      } else {
#pragma unroll
        for (int e = 0; e < 8; ++e) {
          const float xg = acc[i][j][e] + bj;
          outB[(size_t)(m0 + e) * NN + n] =
              (__bf16)(0.5f * xg * (1.0f + erff(xg * 0.70710678118654752f)));
        }
      }
    }
  }
}

// ---------- flash attention: one wave handles 16 queries of one (b,h) ----------
// Computes S^T = K_blk x Q^T so softmax stats are per-lane (query in lane dim),
// and exp(S^T) is directly the B-fragment for O^T = V^T x P^T.  No LDS needed.
__global__ void __launch_bounds__(32)
attn_kernel(const bf16_t* __restrict__ Q, const bf16_t* __restrict__ Kp,
            const bf16_t* __restrict__ Vt, bf16_t* __restrict__ ctx) {
  const int lane = threadIdx.x;
  const int q = lane & 15, half = lane >> 4;
  const int bid = blockIdx.x;
  const int qt = bid & 63;
  const int h  = (bid >> 6) & 7;
  const int b  = bid >> 9;
  const float scale = 0.125f;  // 1/sqrt(64)

  const bf16_t* qrow = Q + ((size_t)b * N_ + qt * 16 + q) * D_ + h * DH_;
  const v16bf qf0 = load_frag(qrow, half);        // d = 0..31
  const v16bf qf1 = load_frag(qrow + 32, half);   // d = 32..63

  // hoisted per-lane base pointers (advance by constant per k-block)
  const bf16_t* krow = Kp + ((size_t)b * N_ + q) * D_ + h * DH_;          // + kb*D_
  const bf16_t* vrow = Vt + ((size_t)(b * H_ + h) * DH_ + q) * N_;        // + t*16*N_ + kb

  v8f o[4];
#pragma unroll
  for (int t = 0; t < 4; ++t)
#pragma unroll
    for (int e = 0; e < 8; ++e) o[t][e] = 0.f;
  float m_old = -1e30f, l = 0.f;

  for (int kb = 0; kb < N_; kb += 32) {
    // ---- S^T = K_blk x Q^T : load all 4 K-fragments, then 4 WMMAs ----
    const bf16_t* k0p = krow + (size_t)kb * D_;
    v16bf kf[4];
    kf[0] = load_frag(k0p, half);
    kf[1] = load_frag(k0p + 32, half);
    kf[2] = load_frag(k0p + (size_t)16 * D_, half);
    kf[3] = load_frag(k0p + (size_t)16 * D_ + 32, half);
    v8f s0, s1;
#pragma unroll
    for (int e = 0; e < 8; ++e) { s0[e] = 0.f; s1[e] = 0.f; }
    s0 = wmma_bf16(kf[0], qf0, s0);
    s0 = wmma_bf16(kf[1], qf1, s0);
    s1 = wmma_bf16(kf[2], qf0, s1);
    s1 = wmma_bf16(kf[3], qf1, s1);

    // ---- prefetch the 4 V-fragments for this k-block ----
    v16bf vf[4];
#pragma unroll
    for (int t = 0; t < 4; ++t)
      vf[t] = load_frag(vrow + (size_t)t * 16 * N_ + kb, half);

    // ---- per-query (per-lane) online softmax; merge halves with one xor-16 shuffle ----
    float mp = -1e30f;
#pragma unroll
    for (int e = 0; e < 8; ++e) {
      s0[e] *= scale; s1[e] *= scale;
      mp = fmaxf(mp, fmaxf(s0[e], s1[e]));
    }
    mp = fmaxf(mp, __shfl_xor(mp, 16, 32));
    const float m_new = fmaxf(m_old, mp);
    const float alpha = __expf(m_old - m_new);

    union { v16bf v; __bf16 e[16]; } pf;
    float rs = 0.f;
#pragma unroll
    for (int e = 0; e < 8; ++e) {
      const float p0 = __expf(s0[e] - m_new);
      const float p1 = __expf(s1[e] - m_new);
      rs += p0 + p1;
      pf.e[e]     = (__bf16)p0;   // keys kb + half*8 + e
      pf.e[e + 8] = (__bf16)p1;   // keys kb + 16 + half*8 + e
    }
    rs += __shfl_xor(rs, 16, 32);
    l = l * alpha + rs;

    // ---- O^T = V^T x P^T, rescaled online ----
#pragma unroll
    for (int t = 0; t < 4; ++t) {
#pragma unroll
      for (int e = 0; e < 8; ++e) o[t][e] *= alpha;
      o[t] = wmma_bf16(vf[t], pf.v, o[t]);
    }
    m_old = m_new;
  }

  const float inv = 1.0f / l;
  bf16_t* crow = ctx + ((size_t)b * N_ + qt * 16 + q) * D_ + h * DH_;
#pragma unroll
  for (int t = 0; t < 4; ++t) {
    union { v8bf v; __bf16 e[8]; } pk;
#pragma unroll
    for (int e = 0; e < 8; ++e) pk.e[e] = (__bf16)(o[t][e] * inv);
    *reinterpret_cast<v8bf*>(crow + t * 16 + 8 * half) = pk.v;
  }
}

// ---------- host launcher ----------
extern "C" void kernel_launch(void* const* d_in, const int* in_sizes, int n_in,
                              void* d_out, int out_size, void* d_ws, size_t ws_size,
                              hipStream_t stream) {
  (void)in_sizes; (void)n_in; (void)out_size; (void)ws_size;
  const float* tokens = (const float*)d_in[0];
  const float* pe     = (const float*)d_in[1];
  const float* ln1_g  = (const float*)d_in[2];
  const float* ln1_b  = (const float*)d_in[3];
  const float* Wq     = (const float*)d_in[4];
  const float* bq     = (const float*)d_in[5];
  const float* Wk     = (const float*)d_in[6];
  const float* bk     = (const float*)d_in[7];
  const float* Wv     = (const float*)d_in[8];
  const float* bv     = (const float*)d_in[9];
  const float* Wp     = (const float*)d_in[10];
  const float* bp     = (const float*)d_in[11];
  const float* ln2_g  = (const float*)d_in[12];
  const float* ln2_b  = (const float*)d_in[13];
  const float* W1     = (const float*)d_in[14];
  const float* b1     = (const float*)d_in[15];
  const float* W2     = (const float*)d_in[16];
  const float* b2     = (const float*)d_in[17];
  float* out = (float*)d_out;

  // workspace carve-up (all sizes multiples of 256B)
  char* ws = (char*)d_ws;
  const size_t SZ_ACT_BF = (size_t)MTOT * D_ * 2;    //  8 MB
  const size_t SZ_ACT_F  = (size_t)MTOT * D_ * 4;    // 16 MB
  const size_t SZ_H_BF   = (size_t)MTOT * DFF_ * 2;  // 32 MB
  const size_t SZ_WDD    = (size_t)D_ * D_ * 2;
  const size_t SZ_WDF    = (size_t)D_ * DFF_ * 2;
  bf16_t* x2   = (bf16_t*)ws;              ws += SZ_ACT_BF;
  bf16_t* qkin = (bf16_t*)ws;              ws += SZ_ACT_BF;
  bf16_t* Qb   = (bf16_t*)ws;              ws += SZ_ACT_BF;
  bf16_t* Kb   = (bf16_t*)ws;              ws += SZ_ACT_BF;
  bf16_t* Vtb  = (bf16_t*)ws;              ws += SZ_ACT_BF;
  bf16_t* ctx  = (bf16_t*)ws;              ws += SZ_ACT_BF;
  bf16_t* x2b  = (bf16_t*)ws;              ws += SZ_ACT_BF;
  bf16_t* hbuf = (bf16_t*)ws;              ws += SZ_H_BF;
  float*  tok1 = (float*)ws;               ws += SZ_ACT_F;
  bf16_t* WqT  = (bf16_t*)ws;              ws += SZ_WDD;
  bf16_t* WkT  = (bf16_t*)ws;              ws += SZ_WDD;
  bf16_t* WvT  = (bf16_t*)ws;              ws += SZ_WDD;
  bf16_t* WpT  = (bf16_t*)ws;              ws += SZ_WDD;
  bf16_t* W1T  = (bf16_t*)ws;              ws += SZ_WDF;
  bf16_t* W2T  = (bf16_t*)ws;              ws += SZ_WDF;

  // 1) weights -> transposed bf16
  {
    const int nDD = D_ * D_, nDF = D_ * DFF_;
    wtrans_kernel<<<(nDD + 255) / 256, 256, 0, stream>>>(Wq, WqT, D_, D_);
    wtrans_kernel<<<(nDD + 255) / 256, 256, 0, stream>>>(Wk, WkT, D_, D_);
    wtrans_kernel<<<(nDD + 255) / 256, 256, 0, stream>>>(Wv, WvT, D_, D_);
    wtrans_kernel<<<(nDD + 255) / 256, 256, 0, stream>>>(Wp, WpT, D_, D_);
    wtrans_kernel<<<(nDF + 255) / 256, 256, 0, stream>>>(W1, W1T, D_, DFF_);   // Wt[2048][512]
    wtrans_kernel<<<(nDF + 255) / 256, 256, 0, stream>>>(W2, W2T, DFF_, D_);   // Wt[512][2048]
  }

  // 2) LN1 -> x2 (bf16), qkin = x2 + PE (bf16)
  ln_kernel<true><<<MTOT / 4, 128, 0, stream>>>(tokens, pe, ln1_g, ln1_b, x2, qkin);

  // 3-5) Q, K, V projections
  dim3 gDD(D_ / 64, MTOT / 64);
  gemm_wmma<D_, D_, 0><<<gDD, 32, 0, stream>>>(qkin, WqT, bq, nullptr, Qb, nullptr);
  gemm_wmma<D_, D_, 0><<<gDD, 32, 0, stream>>>(qkin, WkT, bk, nullptr, Kb, nullptr);
  gemm_wmma<D_, D_, 1><<<gDD, 32, 0, stream>>>(x2,   WvT, bv, nullptr, Vtb, nullptr);

  // 6) attention: B*H*(N/16) waves
  attn_kernel<<<B_ * H_ * (N_ / 16), 32, 0, stream>>>(Qb, Kb, Vtb, ctx);

  // 7) output projection + residual -> tok1 (f32)
  gemm_wmma<D_, D_, 2><<<gDD, 32, 0, stream>>>(ctx, WpT, bp, tokens, nullptr, tok1);

  // 8) LN2 -> x2b (bf16)
  ln_kernel<false><<<MTOT / 4, 128, 0, stream>>>(tok1, nullptr, ln2_g, ln2_b, x2b, nullptr);

  // 9) MLP1 + exact GELU -> hbuf (bf16)
  dim3 gDF(DFF_ / 64, MTOT / 64);
  gemm_wmma<DFF_, D_, 3><<<gDF, 32, 0, stream>>>(x2b, W1T, b1, nullptr, hbuf, nullptr);

  // 10) MLP2 + residual -> d_out (f32)
  gemm_wmma<D_, DFF_, 2><<<gDD, 32, 0, stream>>>(hbuf, W2T, b2, tok1, nullptr, out);
}
